// FeatureAttention_2757369004055
// MI455X (gfx1250) — compile-verified, matrix operand order checked by reading, not run on hardware
//
#include <hip/hip_runtime.h>
#include <math.h>

// Problem constants (match reference)
#define NROWS   100000
#define DIM     512
#define HID     256
#define NTILE   (NROWS / 16)     // 6250 row tiles (exact: 100000 = 16*6250)
#define KSTEPS  (DIM / 32)       // 16 K-blocks of 32
#define NTL     (HID / 16)       // 16 N-tiles of 16

#define NWG     256              // persistent workgroups
#define TPB     512              // 16 waves per WG
#define WAVES_TOTAL (NWG * (TPB / 32))

typedef __attribute__((ext_vector_type(16))) __bf16 v16bf;
typedef __attribute__((ext_vector_type(8)))  float  v8f;

// Full W1 (512x256 fp32, row-major k x n) converted to bf16 and stored in LDS
// pre-swizzled into per-lane WMMA B-fragments:
//   ldsB[(((kb*16 + nt)*32 + lane)*16) + e]
//   lane = (n & 15) + 16 * ((ksub >> 3) & 1)      (ksub = k & 31)
//   e    = (ksub & 7) + ((ksub & 16) ? 8 : 0)
// i.e. lanes 0-15 hold K {0..7,16..23}, lanes 16-31 hold K {8..15,24..31},
// matching the ISA 16-bit operand layout (ISA 7.12.2). Each B fragment is then
// two ds_load_b128 per lane. Total: 16*16*32*16 bf16 = 256 KB (<320 KB WGP LDS).
__device__ __forceinline__ void stage_w1_full(const float* __restrict__ W1,
                                              __bf16* lds, int tid) {
  const float4* src = (const float4*)W1;
#pragma unroll 4
  for (int i = 0; i < 64; ++i) {
    int    c = tid + i * TPB;        // float4 chunk 0..32767 (coalesced)
    float4 v = src[c];
    int lin  = c * 4;                // linear index into [512][256] fp32
    int k    = lin >> 8;             // 0..511
    int n0   = lin & 255;
    int kb   = k >> 5;
    int ksub = k & 31;
    int grp  = (ksub >> 3) & 1;
    int e    = (ksub & 7) + ((ksub & 16) ? 8 : 0);
    float f[4] = {v.x, v.y, v.z, v.w};
#pragma unroll
    for (int j = 0; j < 4; ++j) {
      int n    = n0 + j;
      int nt   = n >> 4;
      int lane = (n & 15) + grp * 16;
      lds[((((kb * NTL) + nt) * 32 + lane) * 16) + e] = (__bf16)f[j];
    }
  }
}

// One 16-row MLP scorer pass. On return, every lane of each 16-lane half holds
// in xr[r] the pre-activation score for row r + 8*(lane>=16) of the tile.
__device__ __forceinline__ void mlp_rows16(const float* __restrict__ z,
                                           const float* __restrict__ b1,
                                           const float* __restrict__ W2,
                                           const __bf16* lds, int row0,
                                           float xr[8]) {
  const int lane = threadIdx.x & 31;
  const int m    = lane & 15;        // A-matrix row within tile
  const int hi   = lane >> 4;        // lane half selects K sub-ranges

  v8f acc[NTL];
  v8f zero = {};
#pragma unroll
  for (int nt = 0; nt < NTL; ++nt) acc[nt] = zero;

  for (int kb = 0; kb < KSTEPS; ++kb) {
    // A fragment (16x32 bf16) straight from global fp32 z rows.
    const float* zr = z + (size_t)(row0 + m) * DIM + kb * 32 + hi * 8;
    float4 f0 = *(const float4*)(zr + 0);
    float4 f1 = *(const float4*)(zr + 4);
    float4 f2 = *(const float4*)(zr + 16);
    float4 f3 = *(const float4*)(zr + 20);
    v16bf a;
    a[0]  = (__bf16)f0.x; a[1]  = (__bf16)f0.y; a[2]  = (__bf16)f0.z; a[3]  = (__bf16)f0.w;
    a[4]  = (__bf16)f1.x; a[5]  = (__bf16)f1.y; a[6]  = (__bf16)f1.z; a[7]  = (__bf16)f1.w;
    a[8]  = (__bf16)f2.x; a[9]  = (__bf16)f2.y; a[10] = (__bf16)f2.z; a[11] = (__bf16)f2.w;
    a[12] = (__bf16)f3.x; a[13] = (__bf16)f3.y; a[14] = (__bf16)f3.z; a[15] = (__bf16)f3.w;

    const __bf16* bbase = lds + (size_t)kb * (NTL * 32 * 16) + (size_t)lane * 16;
#pragma unroll
    for (int nt = 0; nt < NTL; ++nt) {
      v16bf b = *(const v16bf*)(bbase + (size_t)nt * (32 * 16));
      acc[nt] = __builtin_amdgcn_wmma_f32_16x16x32_bf16(
          false, a, false, b, (short)0, acc[nt], false, false);
    }
  }

  // h = relu(acc + b1); partial dot with W2 over this lane's columns.
#pragma unroll
  for (int r = 0; r < 8; ++r) xr[r] = 0.f;
#pragma unroll
  for (int nt = 0; nt < NTL; ++nt) {
    int   n   = nt * 16 + (lane & 15);
    float b1n = b1[n];
    float w2n = W2[n];
#pragma unroll
    for (int r = 0; r < 8; ++r) {
      float h = acc[nt][r] + b1n;
      h = fmaxf(h, 0.f);
      xr[r] = fmaf(h, w2n, xr[r]);
    }
  }
  // Reduce across the 16 lanes of each half-wave (C layout: half = M/8).
#pragma unroll
  for (int r = 0; r < 8; ++r) {
    float v = xr[r];
    v += __shfl_xor(v, 1, 32);
    v += __shfl_xor(v, 2, 32);
    v += __shfl_xor(v, 4, 32);
    v += __shfl_xor(v, 8, 32);
    xr[r] = v;
  }
}

__global__ __launch_bounds__(TPB) void featattn_kernel(
    const float* __restrict__ z1, const float* __restrict__ z2,
    const float* __restrict__ W1, const float* __restrict__ b1,
    const float* __restrict__ W2, const float* __restrict__ b2,
    float* __restrict__ out) {
  __shared__ __align__(32) __bf16 ldsB[KSTEPS * NTL * 32 * 16];   // 256 KB

  const int tid  = threadIdx.x;
  const int lane = tid & 31;
  const int wave = tid >> 5;
  (void)b2;  // b2 cancels in sigmoid(x - y); output never uses raw scores

  // One-time W1 fp32 -> bf16 fragment swizzle into LDS, then no more barriers.
  stage_w1_full(W1, ldsB, tid);
  __syncthreads();

  const int gwave = blockIdx.x * (TPB / 32) + wave;

  for (int tile = gwave; tile < NTILE; tile += WAVES_TOTAL) {
    const int row0 = tile * 16;

    float xr[8], yr[8];
    mlp_rows16(z1, b1, W2, ldsB, row0, xr);
    mlp_rows16(z2, b1, W2, ldsB, row0, yr);

    // score_x = e^x / (e^x + e^y) == sigmoid(x - y)
    float s[8];
#pragma unroll
    for (int r = 0; r < 8; ++r)
      s[r] = 1.f / (1.f + __expf(yr[r] - xr[r]));

    // out = s*z1 + (1-s)*z2, coalesced float4 writes (z re-reads hit L2).
#pragma unroll
    for (int mrow = 0; mrow < 16; ++mrow) {
      float sm = __shfl(s[mrow & 7], (mrow >> 3) * 16, 32);
      const float4* p1 = (const float4*)(z1 + (size_t)(row0 + mrow) * DIM);
      const float4* p2 = (const float4*)(z2 + (size_t)(row0 + mrow) * DIM);
      float4*       po = (float4*)(out + (size_t)(row0 + mrow) * DIM);
#pragma unroll
      for (int j = 0; j < 4; ++j) {
        int    c = lane + j * 32;
        float4 a = p1[c];
        float4 b = p2[c];
        float4 o;
        o.x = fmaf(sm, a.x - b.x, b.x);
        o.y = fmaf(sm, a.y - b.y, b.y);
        o.z = fmaf(sm, a.z - b.z, b.z);
        o.w = fmaf(sm, a.w - b.w, b.w);
        po[c] = o;
      }
    }
  }
}

extern "C" void kernel_launch(void* const* d_in, const int* in_sizes, int n_in,
                              void* d_out, int out_size, void* d_ws, size_t ws_size,
                              hipStream_t stream) {
  (void)in_sizes; (void)n_in; (void)out_size; (void)d_ws; (void)ws_size;
  const float* z1 = (const float*)d_in[0];
  const float* z2 = (const float*)d_in[1];
  const float* W1 = (const float*)d_in[2];
  const float* b1 = (const float*)d_in[3];
  const float* W2 = (const float*)d_in[4];
  const float* b2 = (const float*)d_in[5];
  float* out = (float*)d_out;

  featattn_kernel<<<dim3(NWG), dim3(TPB), 0, stream>>>(z1, z2, W1, b1, W2, b2, out);
}
